// HMM_8186207666604
// MI455X (gfx1250) — matrix operand order repaired and checked by reading, not compiled, not run on hardware
//
#include <hip/hip_runtime.h>
#include <hip/hip_bf16.h>

// ---------------------------------------------------------------------------
// HMM forward (scaled-probability reformulation) for MI455X / gfx1250.
//
//   N=512 states, M=32000 vocab, B=64 batch, TMAX=256.
//
//   Per step:  s[b,j] = sum_k p[b,k] * A[j,k]      <-- bf16 WMMA 16x16x32
//              q[b,j] = s[b,j] * exp(log_E[j,x_t])
//              c[b]  += log(max_j q);  p = q / max_j q
//   Answer at t=T[b]-1:  out[b] = c_old[b] + log(sum_j q[b,j])
//
//   Emission slab for step t+1 is staged into LDS with
//   GLOBAL_LOAD_ASYNC_TO_LDS_B128 (ASYNCcnt) while step t's WMMAs run.
//
// Workspace layout (<= ~33 MB):
//   [0)            A_packed  bf16 [512][512]  row-major A[j][k]=softmax_j(trans[:,k])
//   [524288)       lse_emis  f32  [512]
//   [526336)       log_pi    f32  [512]
//   [528384)       em        f32  [TMAX][B][N]  log_E[j, x[b,t]]
// ---------------------------------------------------------------------------

typedef __bf16 bf16;
typedef __attribute__((ext_vector_type(8)))  __bf16 v8bf;
typedef __attribute__((ext_vector_type(16))) __bf16 v16bf;
typedef __attribute__((ext_vector_type(8)))  float  v8f;

#define HN    512
#define HM    32000
#define HB    64
#define HT    256
#define BPW   16      // batches per workgroup
#define PSTR  520     // p-buffer LDS row stride (bf16): 4-bank rotation/row
#define ESTR  516     // em-buffer LDS row stride (f32): rows m and m+8 differ by 32 banks

static __device__ __forceinline__ v16bf bfcat(v8bf lo, v8bf hi) {
    return __builtin_shufflevector(lo, hi, 0,1,2,3,4,5,6,7,8,9,10,11,12,13,14,15);
}

// LDS byte offset of a generic pointer that points into LDS
static __device__ __forceinline__ unsigned lds_off(const void* p) {
    return (unsigned)(unsigned long long)
           (const __attribute__((address_space(3))) char*)p;
}

// ----------------------- preprocessing kernels -----------------------------

// A_packed[j*N + k] = softmax over j of trans[j*N + k]  (column-wise), bf16.
__global__ void hmm_trans_kernel(const float* __restrict__ trans,
                                 bf16* __restrict__ Ap) {
    int k = blockIdx.x * blockDim.x + threadIdx.x;
    if (k >= HN) return;
    float mx = -INFINITY;
    for (int j = 0; j < HN; ++j) mx = fmaxf(mx, trans[j * HN + k]);
    float s = 0.f;
    for (int j = 0; j < HN; ++j) s += expf(trans[j * HN + k] - mx);
    float inv = 1.0f / s;
    for (int j = 0; j < HN; ++j)
        Ap[(size_t)j * HN + k] = (bf16)(expf(trans[j * HN + k] - mx) * inv);
}

// log_pi = log_softmax(prior)
__global__ void hmm_prior_kernel(const float* __restrict__ prior,
                                 float* __restrict__ logpi) {
    __shared__ float red[HN];
    int t = threadIdx.x;
    float v = prior[t];
    red[t] = v;
    __syncthreads();
    for (int s = HN / 2; s > 0; s >>= 1) {
        if (t < s) red[t] = fmaxf(red[t], red[t + s]);
        __syncthreads();
    }
    float mx = red[0];
    __syncthreads();
    red[t] = expf(v - mx);
    __syncthreads();
    for (int s = HN / 2; s > 0; s >>= 1) {
        if (t < s) red[t] += red[t + s];
        __syncthreads();
    }
    logpi[t] = v - (mx + logf(red[0]));
}

// lse[j] = logsumexp over m of emis[j, :]
__global__ void hmm_lse_kernel(const float* __restrict__ emis,
                               float* __restrict__ lse) {
    __shared__ float rm[256], rs[256];
    int j = blockIdx.x, t = threadIdx.x;
    const float* row = emis + (size_t)j * HM;
    float mx = -INFINITY, s = 0.f;
    for (int i = t; i < HM; i += 256) {
        float v = row[i];
        if (v > mx) { s = s * expf(mx - v) + 1.0f; mx = v; }
        else          s += expf(v - mx);
    }
    rm[t] = mx; rs[t] = s;
    __syncthreads();
    for (int w = 128; w > 0; w >>= 1) {
        if (t < w) {
            float m1 = rm[t], s1 = rs[t], m2 = rm[t + w], s2 = rs[t + w];
            float m = fmaxf(m1, m2);
            rm[t] = m;
            rs[t] = s1 * expf(m1 - m) + s2 * expf(m2 - m);
        }
        __syncthreads();
    }
    if (t == 0) lse[j] = rm[0] + logf(rs[0]);
}

// em[t][b][j] = emis[j, x[b,t]] - lse[j]
__global__ void hmm_gather_kernel(const float* __restrict__ emis,
                                  const float* __restrict__ lse,
                                  const int* __restrict__ x,
                                  float* __restrict__ em) {
    int bt = blockIdx.x;
    int b = bt >> 8;          // TMAX == 256
    int t = bt & 255;
    int j = threadIdx.x;
    int xv = x[b * HT + t];
    em[((size_t)t * HB + b) * HN + j] = emis[(size_t)j * HM + xv] - lse[j];
}

// ----------------------------- forward kernel ------------------------------

// Issue async copies: 16 rows x 512 f32 (global, row stride HN) -> LDS rows
// of ESTR f32.  256 threads, 8 x b128 per thread, tracked by ASYNCcnt.
__device__ __forceinline__ void stage_em_async(const float* __restrict__ gsrc,
                                               float* lbase, int tid) {
    const int m = tid >> 4;     // 0..15 : batch row
    const int c = tid & 15;     // 0..15 : 16B chunk id
    const char* g = (const char*)(gsrc + (size_t)m * HN) + c * 16;
    unsigned    l = lds_off(lbase + (size_t)m * ESTR) + c * 16;
    #pragma unroll
    for (int i = 0; i < 8; ++i) {
        asm volatile("global_load_async_to_lds_b128 %0, %1, off"
                     :: "v"(l + i * 256), "v"(g + i * 256) : "memory");
    }
}

__device__ __forceinline__ void step_epilogue(
    float (*q)[8], bf16* Pw, int* RMaxp, float* RSump, float* Cs,
    const int* Tl, float* out, int b0, int t, int tid, int ln, int hi,
    const int* jt)
{
    // half-wave shuffle reduce over the 16 j-columns, then LDS atomics
    #pragma unroll
    for (int r = 0; r < 8; ++r) {
        int m = r + hi * 8;
        float mx = q[0][r], sm = q[0][r];
        #pragma unroll
        for (int i = 1; i < 4; ++i) { mx = fmaxf(mx, q[i][r]); sm += q[i][r]; }
        #pragma unroll
        for (int off = 1; off < 16; off <<= 1) {
            mx = fmaxf(mx, __shfl_xor(mx, off, 32));
            sm += __shfl_xor(sm, off, 32);
        }
        if (ln == 0) {
            atomicMax(&RMaxp[m], __float_as_int(mx));  // q >= 0: int order == float order
            atomicAdd(&RSump[m], sm);
        }
    }
    __syncthreads();   // barrier 1: reductions complete

    if (tid < BPW) {
        int m = tid;
        float mx = __int_as_float(RMaxp[m]);
        float sm = RSump[m];
        float c0 = Cs[m];
        if (t == Tl[m] - 1) out[b0 + m] = c0 + logf(sm);
        Cs[m] = c0 + logf(mx);
    }

    // normalize and store new p (bf16) into this step's ping-pong buffer
    #pragma unroll
    for (int r = 0; r < 8; ++r) {
        int m = r + hi * 8;
        float inv = 1.0f / __int_as_float(RMaxp[m]);
        #pragma unroll
        for (int i = 0; i < 4; ++i)
            Pw[(size_t)m * PSTR + jt[i] + ln] = (bf16)(q[i][r] * inv);
    }

    // publish this step's async em staging before releasing the barrier
    asm volatile("s_wait_asynccnt 0x0" ::: "memory");
    __syncthreads();   // barrier 2: p + staged em ready for next step
}

__global__ __launch_bounds__(256)
void hmm_forward_kernel(const bf16* __restrict__ Ap,
                        const float* __restrict__ em,
                        const float* __restrict__ logpi,
                        const int* __restrict__ Tlen,
                        float* __restrict__ out)
{
    __shared__ __align__(16) bf16  P[2][BPW * PSTR];     // 33.3 KB
    __shared__ __align__(16) float EmB[2][BPW * ESTR];   // 66.0 KB
    __shared__ int   RMax[2][BPW];
    __shared__ float RSum[2][BPW];
    __shared__ float Cs[BPW];
    __shared__ int   Tl[BPW];

    const int tid  = threadIdx.x;
    const int lane = tid & 31;
    const int wave = tid >> 5;
    const int ln   = lane & 15;
    const int hi   = lane >> 4;            // 0 for lanes 0-15, 1 for 16-31
    const int b0   = blockIdx.x * BPW;

    if (tid < BPW) {
        RMax[0][tid] = 0; RMax[1][tid] = 0;
        RSum[0][tid] = 0.f; RSum[1][tid] = 0.f;
        Cs[tid] = 0.f;
        Tl[tid] = Tlen[b0 + tid];
    }
    __syncthreads();

    int jt[4];
    #pragma unroll
    for (int i = 0; i < 4; ++i) jt[i] = wave * 64 + i * 16;

    float q[4][8];

    // -------- t = 0 : q = exp(log_pi + em0), em0 straight from global --------
    {
        // start staging the t=1 emission slab into EmB[1]
        stage_em_async(em + (size_t)HB * HN + (size_t)b0 * HN, &EmB[1][0], tid);

        const float* em0 = em + (size_t)b0 * HN;
        #pragma unroll
        for (int i = 0; i < 4; ++i) {
            int j = jt[i] + ln;
            float lp = logpi[j];
            #pragma unroll
            for (int r = 0; r < 8; ++r) {
                int m = r + hi * 8;
                q[i][r] = expf(lp + em0[(size_t)m * HN + j]);
            }
        }
        step_epilogue(q, &P[0][0], &RMax[0][0], &RSum[0][0], Cs, Tl, out,
                      b0, 0, tid, ln, hi, jt);
    }

    // -------- t = 1 .. TMAX-1 : WMMA matmul steps --------
    for (int t = 1; t < HT; ++t) {
        const int par = t & 1;
        if (tid < BPW) { RMax[par ^ 1][tid] = 0; RSum[par ^ 1][tid] = 0.f; }

        // stage em slab for step t+1 into the other em buffer (hidden by WMMAs)
        if (t + 1 < HT)
            stage_em_async(em + ((size_t)(t + 1) * HB + b0) * HN,
                           &EmB[par ^ 1][0], tid);

        const bf16* Prd = &P[par ^ 1][0];
        v8f acc[4] = {};

        #pragma unroll 4
        for (int k0 = 0; k0 < HN; k0 += 32) {
            // A operand: p[m = ln][K], chunks at k0 + 8*hi and +16 (ISA layout)
            const v8bf* pa = (const v8bf*)(Prd + ln * PSTR + k0 + hi * 8);
            v16bf av = bfcat(pa[0], pa[2]);
            #pragma unroll
            for (int i = 0; i < 4; ++i) {
                // B operand: column j = jt[i]+ln, chunks at k0 + 8*hi and +16 (ISA layout)
                const v8bf* pb = (const v8bf*)(Ap + (size_t)(jt[i] + ln) * HN
                                               + k0 + hi * 8);
                v16bf bv = bfcat(pb[0], pb[2]);
                acc[i] = __builtin_amdgcn_wmma_f32_16x16x32_bf16(
                    false, av, false, bv, (short)0, acc[i], false, false);
            }
        }

        // emission from the LDS-staged slab (loaded during step t-1)
        const float* emt = &EmB[par][0];
        #pragma unroll
        for (int i = 0; i < 4; ++i) {
            int j = jt[i] + ln;
            #pragma unroll
            for (int r = 0; r < 8; ++r) {
                int m = r + hi * 8;
                q[i][r] = acc[i][r] * expf(emt[m * ESTR + j]);
            }
        }

        step_epilogue(q, &P[par][0], &RMax[par][0], &RSum[par][0], Cs, Tl, out,
                      b0, t, tid, ln, hi, jt);
    }
}

// ------------------------------- launcher ----------------------------------

extern "C" void kernel_launch(void* const* d_in, const int* in_sizes, int n_in,
                              void* d_out, int out_size, void* d_ws, size_t ws_size,
                              hipStream_t stream) {
    const int*   x     = (const int*)d_in[0];
    const int*   T     = (const int*)d_in[1];
    const float* trans = (const float*)d_in[2];
    const float* emis  = (const float*)d_in[3];
    const float* prior = (const float*)d_in[4];
    float*       out   = (float*)d_out;

    char* ws = (char*)d_ws;
    bf16*  Ap    = (bf16*)ws;                            // 512 KB
    float* lse   = (float*)(ws + 524288);                // 2 KB
    float* logpi = (float*)(ws + 524288 + 2048);         // 2 KB
    float* em    = (float*)(ws + 524288 + 4096);         // 32 MB

    hmm_trans_kernel <<<2, 256, 0, stream>>>(trans, Ap);
    hmm_prior_kernel <<<1, HN, 0, stream>>>(prior, logpi);
    hmm_lse_kernel   <<<HN, 256, 0, stream>>>(emis, lse);
    hmm_gather_kernel<<<HB * HT, HN, 0, stream>>>(emis, lse, x, em);
    hmm_forward_kernel<<<HB / BPW, 256, 0, stream>>>(Ap, em, logpi, T, out);
}
